// PhaseModel_2396591751776
// MI455X (gfx1250) — compile-verified, matrix operand order checked by reading, not run on hardware
//
#include <hip/hip_runtime.h>

typedef __attribute__((ext_vector_type(16))) _Float16 v16h;
typedef __attribute__((ext_vector_type(8)))  float    v8f;

#define N_NODES 30000
#define N_EDGES 300000

// 16-bit A-fragment / B-fragment K mapping (16x32 tile, wave32):
// vgpr pair v (0..7), half-wave hf: k0 = 16*(v>>2) + 8*hf + 2*(v&3)
__device__ __forceinline__ int kmap(int v, int hf) {
    return 16 * (v >> 2) + 8 * hf + 2 * (v & 3);
}

// ---------------------------------------------------------------- utilities
__global__ void zero_f32(float* __restrict__ p, int n) {
    int i = blockIdx.x * blockDim.x + threadIdx.x;
    if (i < n) p[i] = 0.f;
}

// ---- pre-swizzle W (fp32 [K, NK]) into per-lane v16h B-fragments, fp16,
// ---- ntiles N-tiles x kfrags K-fragments (K padded to kfrags*32, N to ntiles*16)
__global__ void pack_Bk(const float* __restrict__ W, _Float16* __restrict__ bp,
                        int NK, int ntiles, int K, int kfrags)
{
    int g = blockIdx.x * blockDim.x + threadIdx.x;
    if (g >= ntiles * kfrags * 32) return;
    int L  = g & 31;
    int kf = (g >> 5) % kfrags;
    int t  = g / (32 * kfrags);
    int n = L & 15, hf = L >> 4;
    int col = t * 16 + n;
    _Float16* dst = bp + (size_t)g * 16;
#pragma unroll
    for (int v = 0; v < 8; v++) {
#pragma unroll
        for (int s = 0; s < 2; s++) {
            int k = kf * 32 + kmap(v, hf) + s;
            float val = (k < K && col < NK) ? W[k * NK + col] : 0.f;
            dst[2 * v + s] = (_Float16)val;
        }
    }
}

// =====================================================================
// Fully fused ECC layer: one wave32 per 16 edges.
//   h1 = relu(e*W1+b1)            -> built directly in A-fragment registers
//   h2 = relu(h1@W2+b2)           -> 4 WMMA tiles (K=30->32), LDS transpose
//   h3 = relu(h2@W3+b3)           -> 2 N-tiles x 2 K-frags chained WMMA (K=60->64)
//   kern = h3@W4+b4 (per tile)    -> WMMA, contracted against x[src] in LDS
//   accum[tgt] += msg             -> global fp32 atomics
// kern / h2 / h3 never touch HBM.
// =====================================================================
template <int FIN, int FOUT, int NT>
__global__ __launch_bounds__(32) void ecc_layer(
    const float* __restrict__ e,
    const float* __restrict__ W1, const float* __restrict__ B1,
    const _Float16* __restrict__ W2p, const float* __restrict__ B2,
    const _Float16* __restrict__ W3p, const float* __restrict__ B3,
    const _Float16* __restrict__ W4p, const float* __restrict__ B4,
    const float* __restrict__ xin,
    const int* __restrict__ tgt, const int* __restrict__ src,
    float* __restrict__ accum)
{
    __shared__ float    sW1[30], sB1[30];
    __shared__ _Float16 sh2[16 * 64];
    __shared__ _Float16 sh3[16 * 32];
    __shared__ float    smsg[16 * FOUT];
    __shared__ float    sx[16 * FIN];
    __shared__ int      stgt[16];

    int L = threadIdx.x;
    int m = L & 15, hf = L >> 4;
    int edge = blockIdx.x * 16 + m;          // E % 16 == 0: no partial tiles

    if (L < 30) { sW1[L] = W1[L]; sB1[L] = B1[L]; }
    for (int i = L; i < 16 * FOUT; i += 32) smsg[i] = 0.f;
    if (hf == 0) stgt[m] = tgt[edge];
    int sn = src[edge];
    for (int i = hf; i < FIN; i += 2) sx[m * FIN + i] = xin[(size_t)sn * FIN + i];
    float ev = e[edge];
    __syncthreads();

    // ---- stage 1: h1 in A-fragment layout, registers only (K=30 padded to 32)
    v16h a2;
#pragma unroll
    for (int v = 0; v < 8; v++) {
#pragma unroll
        for (int s = 0; s < 2; s++) {
            int k = kmap(v, hf) + s;
            float h = 0.f;
            if (k < 30) {
                h = ev * sW1[k] + sB1[k];
                h = h > 0.f ? h : 0.f;
            }
            a2[2 * v + s] = (_Float16)h;
        }
    }

    // ---- stage 2: h2 = relu(h1 @ W2 + b2), 4 N-tiles of 16 (cols 0..63, 60+ zero)
#pragma unroll
    for (int t = 0; t < 4; t++) {
        v16h b = *(const v16h*)(W2p + (size_t)(t * 32 + L) * 16);
        v8f c = {};
        c = __builtin_amdgcn_wmma_f32_16x16x32_f16(false, a2, false, b,
                                                   (short)0, c, false, false);
        int col = t * 16 + m;                // D column index == lane%16
        float bb = (col < 60) ? B2[col] : 0.f;
#pragma unroll
        for (int v = 0; v < 8; v++) {
            float h = c[v] + bb;
            h = (col < 60 && h > 0.f) ? h : 0.f;
            sh2[(v + 8 * hf) * 64 + col] = (_Float16)h;   // C-layout -> row-major LDS
        }
    }
    __syncthreads();

    // ---- LDS transpose: A-fragments for K=0..31 and K=32..63
    v16h a3a, a3b;
#pragma unroll
    for (int v = 0; v < 8; v++) {
        int k0 = kmap(v, hf);
        a3a[2 * v]     = sh2[m * 64 + k0];
        a3a[2 * v + 1] = sh2[m * 64 + k0 + 1];
        a3b[2 * v]     = sh2[m * 64 + 32 + k0];
        a3b[2 * v + 1] = sh2[m * 64 + 32 + k0 + 1];
    }

    // ---- stage 3: h3 = relu(h2 @ W3 + b3), 2 N-tiles, K=64 via chained WMMA
#pragma unroll
    for (int t = 0; t < 2; t++) {
        v16h b0 = *(const v16h*)(W3p + (size_t)((t * 2 + 0) * 32 + L) * 16);
        v16h b1 = *(const v16h*)(W3p + (size_t)((t * 2 + 1) * 32 + L) * 16);
        v8f c = {};
        c = __builtin_amdgcn_wmma_f32_16x16x32_f16(false, a3a, false, b0,
                                                   (short)0, c, false, false);
        c = __builtin_amdgcn_wmma_f32_16x16x32_f16(false, a3b, false, b1,
                                                   (short)0, c, false, false);
        int col = t * 16 + m;
        float bb = (col < 30) ? B3[col] : 0.f;
#pragma unroll
        for (int v = 0; v < 8; v++) {
            float h = c[v] + bb;
            h = (col < 30 && h > 0.f) ? h : 0.f;
            sh3[(v + 8 * hf) * 32 + col] = (_Float16)h;
        }
    }
    __syncthreads();

    // ---- A-fragment for the kern GEMM (K=30 padded to 32)
    v16h a4;
#pragma unroll
    for (int v = 0; v < 8; v++) {
        int k0 = kmap(v, hf);
        a4[2 * v]     = sh3[m * 32 + k0];
        a4[2 * v + 1] = sh3[m * 32 + k0 + 1];
    }

    // ---- stage 4: kern tiles + contraction against gathered x[src]
#pragma unroll 1
    for (int t = 0; t < NT; t++) {
        v16h b = *(const v16h*)(W4p + (size_t)(t * 32 + L) * 16);
        v8f c = {};
        c = __builtin_amdgcn_wmma_f32_16x16x32_f16(false, a4, false, b,
                                                   (short)0, c, false, false);
        int col = t * 16 + m;                // same (i,o) for all 8 acc regs
        if (col < FIN * FOUT) {
            int i = col / FOUT;              // constexpr FOUT -> mul/shift
            int o = col - i * FOUT;
            float bb = B4[col];
#pragma unroll
            for (int v = 0; v < 8; v++) {
                int mr = v + 8 * hf;         // edge row of this accumulator element
                smsg[mr * FOUT + o] += (c[v] + bb) * sx[mr * FIN + i];
            }
        }
    }
    __syncthreads();

    int node = stgt[m];
    for (int o = hf; o < FOUT; o += 2)
        unsafeAtomicAdd(&accum[(size_t)node * FOUT + o], smsg[m * FOUT + o]);
}

// ------------------------------------- x = relu(accum + x @ root + bias), in place
template <int FIN, int FOUT>
__global__ void root_relu(const float* __restrict__ xin, const float* __restrict__ root,
                          const float* __restrict__ bias, float* __restrict__ xout, int n)
{
    __shared__ float sroot[FIN * FOUT];
    __shared__ float sbias[FOUT];
    int t = threadIdx.x;
    for (int i = t; i < FIN * FOUT; i += blockDim.x) sroot[i] = root[i];
    for (int i = t; i < FOUT; i += blockDim.x) sbias[i] = bias[i];
    __syncthreads();
    int node = blockIdx.x * blockDim.x + t;
    if (node >= n) return;
    float xv[FIN];
#pragma unroll
    for (int i = 0; i < FIN; i++) xv[i] = xin[(size_t)node * FIN + i];
    for (int o = 0; o < FOUT; o++) {
        float acc = xout[(size_t)node * FOUT + o] + sbias[o];
#pragma unroll
        for (int i = 0; i < FIN; i++) acc += xv[i] * sroot[i * FOUT + o];
        xout[(size_t)node * FOUT + o] = acc > 0.f ? acc : 0.f;
    }
}

// ------------------------------------------------------------- global sum pool
__global__ void pool_kernel(const float* __restrict__ x, float* __restrict__ pooled,
                            int n, int f)
{
    __shared__ float sp[64];
    int t = threadIdx.x;
    if (t < 64) sp[t] = 0.f;
    __syncthreads();
    int node = blockIdx.x * blockDim.x + t;
    if (node < n)
        for (int o = 0; o < f; o++) atomicAdd(&sp[o], x[(size_t)node * f + o]);
    __syncthreads();
    if (t < f) unsafeAtomicAdd(&pooled[t], sp[t]);
}

// -------------------------------------------------------- final MLP 54->18->6->1
__global__ __launch_bounds__(32) void final_mlp(
    const float* __restrict__ pooled,
    const float* __restrict__ W0, const float* __restrict__ B0,
    const float* __restrict__ W1, const float* __restrict__ B1,
    const float* __restrict__ W2, const float* __restrict__ B2,
    float* __restrict__ out)
{
    if (threadIdx.x != 0) return;
    float h1[18], h2[6];
    for (int o = 0; o < 18; o++) {
        float a = B0[o];
        for (int i = 0; i < 54; i++) a += pooled[i] * W0[i * 18 + o];
        h1[o] = a > 0.f ? a : 0.f;
    }
    for (int o = 0; o < 6; o++) {
        float a = B1[o];
        for (int i = 0; i < 18; i++) a += h1[i] * W1[i * 6 + o];
        h2[o] = a > 0.f ? a : 0.f;
    }
    float a = B2[0];
    for (int i = 0; i < 6; i++) a += h2[i] * W2[i];
    out[0] = a;
}

// ============================================================== host launcher
struct EccP { const float* kW[4]; const float* kb[4]; const float* root; const float* bias; };

template <int FIN, int FOUT>
static void run_layer(const EccP& P, const float* e, const float* xin,
                      const int* tgt, const int* src, float* acc,
                      _Float16* w2p, _Float16* w3p, _Float16* w4p, hipStream_t stream)
{
    constexpr int NT = (FIN * FOUT + 15) / 16;
    zero_f32<<<(N_NODES * FOUT + 255) / 256, 256, 0, stream>>>(acc, N_NODES * FOUT);
    pack_Bk<<<1, 128, 0, stream>>>(P.kW[1], w2p, 60, 4, 30, 1);          // [30,60]
    pack_Bk<<<1, 128, 0, stream>>>(P.kW[2], w3p, 30, 2, 60, 2);          // [60,30], K=64
    pack_Bk<<<(NT * 32 + 255) / 256, 256, 0, stream>>>(P.kW[3], w4p,
                                                       FIN * FOUT, NT, 30, 1);
    ecc_layer<FIN, FOUT, NT><<<N_EDGES / 16, 32, 0, stream>>>(
        e, P.kW[0], P.kb[0], w2p, P.kb[1], w3p, P.kb[2], w4p, P.kb[3],
        xin, tgt, src, acc);
    root_relu<FIN, FOUT><<<(N_NODES + 255) / 256, 256, 0, stream>>>(
        xin, P.root, P.bias, acc, N_NODES);
}

extern "C" void kernel_launch(void* const* d_in, const int* in_sizes, int n_in,
                              void* d_out, int out_size, void* d_ws, size_t ws_size,
                              hipStream_t stream)
{
    (void)out_size; (void)ws_size;
    // ---- top-level inputs (handle insertion-order or sorted-key flattening)
    const float* x; const float* e; const int* ei; int idx;
    if (in_sizes[0] == N_NODES * 2) {           // x, e, edge_index, params...
        x = (const float*)d_in[0];
        e = (const float*)d_in[1];
        ei = (const int*)d_in[2];
        idx = 3;
    } else {                                    // e, edge_index, params..., x
        e = (const float*)d_in[0];
        ei = (const int*)d_in[1];
        idx = 2;
        x = (const float*)d_in[n_in - 1];
    }
    const int* tgt = ei;
    const int* src = ei + N_EDGES;

    // ---- params: ecc layers (detect {bias,knet,root} vs {knet,root,bias} order)
    EccP ecc[3];
    const int fouts[3] = {6, 18, 54};
    for (int l = 0; l < 3; l++) {
        int fout = fouts[l];
        if (in_sizes[idx] == fout) {            // sorted keys: bias first
            ecc[l].bias = (const float*)d_in[idx++];
            for (int j = 0; j < 4; j++) {
                ecc[l].kW[j] = (const float*)d_in[idx++];
                ecc[l].kb[j] = (const float*)d_in[idx++];
            }
            ecc[l].root = (const float*)d_in[idx++];
        } else {                                // insertion order: knet, root, bias
            for (int j = 0; j < 4; j++) {
                ecc[l].kW[j] = (const float*)d_in[idx++];
                ecc[l].kb[j] = (const float*)d_in[idx++];
            }
            ecc[l].root = (const float*)d_in[idx++];
            ecc[l].bias = (const float*)d_in[idx++];
        }
    }
    const float* mlpW[3]; const float* mlpB[3];
    for (int l = 0; l < 3; l++) {
        mlpW[l] = (const float*)d_in[idx++];
        mlpB[l] = (const float*)d_in[idx++];
    }

    // ---- workspace carve
    char* w = (char*)d_ws;
    auto carve = [&](size_t bytes) -> char* {
        char* p = w; w += (bytes + 255) & ~(size_t)255; return p;
    };
    float*    bufA   = (float*)carve((size_t)N_NODES * 64 * 4);
    float*    bufB   = (float*)carve((size_t)N_NODES * 64 * 4);
    float*    bufC   = (float*)carve((size_t)N_NODES * 64 * 4);
    _Float16* w2p    = (_Float16*)carve(4 * 32 * 16 * 2);
    _Float16* w3p    = (_Float16*)carve(4 * 32 * 16 * 2);
    _Float16* w4p    = (_Float16*)carve(64 * 32 * 16 * 2);
    float*    pooled = (float*)carve(256);

    run_layer<2, 6>  (ecc[0], e, x,    tgt, src, bufA, w2p, w3p, w4p, stream);
    run_layer<6, 18> (ecc[1], e, bufA, tgt, src, bufB, w2p, w3p, w4p, stream);
    run_layer<18, 54>(ecc[2], e, bufB, tgt, src, bufC, w2p, w3p, w4p, stream);

    zero_f32<<<1, 64, 0, stream>>>(pooled, 64);
    pool_kernel<<<(N_NODES + 255) / 256, 256, 0, stream>>>(bufC, pooled, N_NODES, 54);
    final_mlp<<<1, 32, 0, stream>>>(pooled, mlpW[0], mlpB[0], mlpW[1], mlpB[1],
                                    mlpW[2], mlpB[2], (float*)d_out);
}
